// SparseTransformerLayer_11063835755126
// MI455X (gfx1250) — compile-verified, matrix operand order checked by reading, not run on hardware
//
#include <hip/hip_runtime.h>
#include <hip/hip_bf16.h>
#include <math.h>

typedef __attribute__((ext_vector_type(16))) _Float16 v16h;
typedef __attribute__((ext_vector_type(8)))  _Float16 v8h;
typedef __attribute__((ext_vector_type(8)))  float    v8f;

#define N_NODES 50000
#define N_EDGES 400000
#define DMODEL  256
#define NHEAD   4
#define HDIM    64

// ---------------------------------------------------------------------------
// helpers
// ---------------------------------------------------------------------------
__device__ __forceinline__ v16h make_v16(v8h lo, v8h hi) {
  v16h a;
#pragma unroll
  for (int i = 0; i < 8; ++i) { a[i] = lo[i]; a[8 + i] = hi[i]; }
  return a;
}

// A fragment (16x32 f16): lane l (g=l>>4, m=l&15) holds
//   halves 0..7  = A[m][k0 + 8g .. k0 + 8g+7]      (contiguous -> ds_load_b128)
//   halves 8..15 = A[m][k0 +16+8g .. k0+16+8g+7]   (contiguous -> ds_load_b128)
__device__ __forceinline__ v16h load_a_frag(const _Float16* ldsA, int ldk, int k0, int lane) {
  const int g = lane >> 4, m = lane & 15;
  const _Float16* p = ldsA + m * ldk + k0 + 8 * g;
  v8h lo = *(const v8h*)p;
  v8h hi = *(const v8h*)(p + 16);
  return make_v16(lo, hi);
}

// B fragment from pre-packed buffer: 32 bytes contiguous per lane, lanes
// contiguous -> two fully coalesced global_load_b128 per wave per k-tile.
__device__ __forceinline__ v16h load_b_frag(const _Float16* Bpack, int KT, int nt, int kt, int lane) {
  const _Float16* p = Bpack + ((size_t)(nt * KT + kt) * 32 + lane) * 16;
  const v8h* q = (const v8h*)p;
  return make_v16(q[0], q[1]);
}

// CDNA5 async global->LDS copy (ASYNCcnt path): 16B per lane, LDS written
// directly by the memory hierarchy (no VGPR round-trip). The wave-relative
// LDS byte offset is the low 32 bits of the generic shared pointer (ISA 10.2).
__device__ __forceinline__ void async_g2l_b128(void* lds_dst, const void* gsrc) {
  unsigned off = (unsigned)(uintptr_t)lds_dst;
  asm volatile("global_load_async_to_lds_b128 %0, %1, off"
               :: "v"(off), "v"(gsrc) : "memory");
}
__device__ __forceinline__ void wait_async0() {
  asm volatile("s_wait_asynccnt 0" ::: "memory");
}

__device__ __forceinline__ unsigned f2key(float f) {
  unsigned u = __float_as_uint(f);
  return (u & 0x80000000u) ? ~u : (u | 0x80000000u);
}
__device__ __forceinline__ float key2f(unsigned k) {
  unsigned u = (k & 0x80000000u) ? (k & 0x7FFFFFFFu) : ~k;
  return __uint_as_float(u);
}

// ---------------------------------------------------------------------------
// elementwise / packing kernels
// ---------------------------------------------------------------------------
__global__ __launch_bounds__(256) void cast_f32_f16_k(const float* __restrict__ s,
                                                      _Float16* __restrict__ d, long long n) {
  long long i = (long long)blockIdx.x * 256 + threadIdx.x;
  if (i < n) d[i] = (_Float16)s[i];
}

__global__ __launch_bounds__(256) void concat_qkv_w_k(const float* __restrict__ Wq,
                                                      const float* __restrict__ Wkv,
                                                      float* __restrict__ Wcat) {
  int idx = blockIdx.x * 256 + threadIdx.x;
  if (idx >= 512 * 768) return;
  int r = idx / 768, c = idx % 768;
  Wcat[idx] = (c < 256) ? Wq[r * 256 + c] : Wkv[r * 512 + (c - 256)];
}

__global__ __launch_bounds__(256) void concat_bias_k(const float* __restrict__ bq,
                                                     const float* __restrict__ bkv,
                                                     float* __restrict__ b768) {
  int i = blockIdx.x * 256 + threadIdx.x;
  if (i >= 768) return;
  b768[i] = (i < 256) ? bq[i] : bkv[i - 256];
}

// Pack row-major KxN f32 weights into WMMA B-fragment order (f16).
__global__ __launch_bounds__(256) void pack_b_k(const float* __restrict__ W,
                                                _Float16* __restrict__ out, int K, int N) {
  int p = blockIdx.x * 256 + threadIdx.x;
  if (p >= K * N) return;
  const int KT = K >> 5;
  int i    = p & 15;
  int lane = (p >> 4) & 31;
  int tile = p >> 9;
  int kt = tile % KT;
  int nt = tile / KT;
  int g  = lane >> 4;
  int n  = nt * 16 + (lane & 15);
  int kk = (i < 8) ? (i + 8 * g) : (16 + (i - 8) + 8 * g);
  int k  = kt * 32 + kk;
  out[p] = (_Float16)W[(size_t)k * N + n];
}

// ---------------------------------------------------------------------------
// Edge GEMM: [hn[src]|he] @ Wq  and  [hn[dst]|he] @ Wkv  fused (768 out cols)
// ---------------------------------------------------------------------------
__global__ __launch_bounds__(256) void edge_qkv_gemm_k(
    const _Float16* __restrict__ hn16, const _Float16* __restrict__ he16,
    const int* __restrict__ srcIdx, const int* __restrict__ dstIdx,
    const _Float16* __restrict__ Bpack, const float* __restrict__ bias768,
    _Float16* __restrict__ Qo, _Float16* __restrict__ Ko, _Float16* __restrict__ Vo) {
  __shared__ _Float16 ldsAq[16 * 520];   // 16 rows x 512 K, stride 520 (conflict-free)
  __shared__ _Float16 ldsAkv[16 * 520];
  __shared__ int sSrc[16], sDst[16];
  const int t = threadIdx.x;
  const int ebase = blockIdx.x * 16;
  if (t < 16) { sSrc[t] = srcIdx[ebase + t]; sDst[t] = dstIdx[ebase + t]; }
  __syncthreads();
  // stage both A tiles via async global->LDS DMA:
  // 2 tiles * 16 rows * 64 b128 chunks = 2048 chunks, 8 per thread
  for (int c = t; c < 2048; c += 256) {
    int tile = c >> 10;
    int row  = (c >> 6) & 15;
    int j    = c & 63;
    const _Float16* sp;
    if (j < 32) {
      int node = tile ? sDst[row] : sSrc[row];
      sp = hn16 + (size_t)node * 256 + j * 8;
    } else {
      sp = he16 + (size_t)(ebase + row) * 256 + (j - 32) * 8;
    }
    _Float16* dp = (tile ? ldsAkv : ldsAq) + row * 520 + j * 8;
    async_g2l_b128(dp, sp);
  }
  wait_async0();
  __syncthreads();
  const int wave = t >> 5, lane = t & 31;
  for (int it = 0; it < 6; ++it) {
    const int nt = it * 8 + wave;                    // 0..47  (768 cols)
    const _Float16* A = (nt < 16) ? ldsAq : ldsAkv;  // wave-uniform
    v8f acc = {};
#pragma unroll
    for (int kt = 0; kt < 16; ++kt) {
      v16h a = load_a_frag(A, 520, kt * 32, lane);
      v16h b = load_b_frag(Bpack, 16, nt, kt, lane);
      acc = __builtin_amdgcn_wmma_f32_16x16x32_f16(false, a, false, b, (short)0, acc,
                                                   false, false);
    }
    const int n = nt * 16 + (lane & 15);
    const float bia = bias768[n];
    const int g = lane >> 4;
    _Float16* out; int col;
    if (n < 256)      { out = Qo; col = n; }
    else if (n < 512) { out = Ko; col = n - 256; }
    else              { out = Vo; col = n - 512; }
#pragma unroll
    for (int r = 0; r < 8; ++r) {
      int row = ebase + r + 8 * g;
      out[(size_t)row * 256 + col] = (_Float16)(acc[r] + bia);
    }
  }
}

// ---------------------------------------------------------------------------
// attention: scores + segment max / expsum / weighted aggregation
// ---------------------------------------------------------------------------
__global__ __launch_bounds__(256) void scores_max_k(
    const _Float16* __restrict__ Q, const _Float16* __restrict__ Kb,
    const int* __restrict__ srcIdx, float* __restrict__ scores,
    unsigned* __restrict__ mkey) {
  int tid = blockIdx.x * 256 + threadIdx.x;
  if (tid >= N_EDGES * NHEAD) return;
  int e = tid >> 2, h = tid & 3;
  const v8h* q = (const v8h*)(Q + (size_t)e * 256 + h * 64);
  const v8h* k = (const v8h*)(Kb + (size_t)e * 256 + h * 64);
  float s = 0.f;
#pragma unroll
  for (int j = 0; j < 8; ++j) {
    v8h qa = q[j], ka = k[j];
#pragma unroll
    for (int x = 0; x < 8; ++x) s += (float)qa[x] * (float)ka[x];
  }
  s *= 0.125f;  // 1/sqrt(64)
  scores[tid] = s;
  atomicMax(mkey + srcIdx[e] * 4 + h, f2key(s));
}

__global__ __launch_bounds__(256) void expsum_k(
    const float* __restrict__ scores, const unsigned* __restrict__ mkey,
    const int* __restrict__ srcIdx, float* __restrict__ ex, float* __restrict__ denom) {
  int tid = blockIdx.x * 256 + threadIdx.x;
  if (tid >= N_EDGES * NHEAD) return;
  int e = tid >> 2, h = tid & 3;
  float m = key2f(mkey[srcIdx[e] * 4 + h]);
  float v = __expf(scores[tid] - m);
  ex[tid] = v;
  atomicAdd(denom + srcIdx[e] * 4 + h, v);
}

__global__ __launch_bounds__(256) void agg_k(
    const float* __restrict__ ex, const float* __restrict__ denom,
    const int* __restrict__ srcIdx, const _Float16* __restrict__ Vb,
    float* __restrict__ agg) {
  int tid = blockIdx.x * 256 + threadIdx.x;
  if (tid >= N_EDGES * 256) return;
  int e = tid >> 8, c = tid & 255, h = c >> 6;
  int s = srcIdx[e];
  float w = ex[e * 4 + h] / denom[s * 4 + h];
  atomicAdd(agg + (size_t)s * 256 + c, w * (float)Vb[tid]);
}

// ---------------------------------------------------------------------------
// generic node-side WMMA GEMM: C(M x N) = A16(M x K) * Bpack + bias, epilogues
// ---------------------------------------------------------------------------
#define MODE_GELU 1
#define MODE_RES  2
__global__ __launch_bounds__(256) void node_gemm_k(
    const _Float16* __restrict__ A16, const _Float16* __restrict__ Bpack,
    const float* __restrict__ bias, const float* __restrict__ residual,
    float* __restrict__ Cf, _Float16* __restrict__ Ch, int N, int K, int mode) {
  __shared__ _Float16 ldsA[16 * 1032];  // supports K up to 1024, stride K+8
  const int t = threadIdx.x;
  const int mbase = blockIdx.x * 16;
  const int ldk = K + 8;
  const int kc = K >> 3;  // b128 chunks per row
  for (int c = t; c < 16 * kc; c += 256) {
    int row = c / kc;
    int j = c - row * kc;
    async_g2l_b128(ldsA + row * ldk + j * 8,
                   A16 + (size_t)(mbase + row) * K + j * 8);
  }
  wait_async0();
  __syncthreads();
  const int wave = t >> 5, lane = t & 31;
  const int KT = K >> 5, NT = N >> 4;
  for (int nt = wave; nt < NT; nt += 8) {
    v8f acc = {};
    for (int kt = 0; kt < KT; ++kt) {
      v16h a = load_a_frag(ldsA, ldk, kt * 32, lane);
      v16h b = load_b_frag(Bpack, KT, nt, kt, lane);
      acc = __builtin_amdgcn_wmma_f32_16x16x32_f16(false, a, false, b, (short)0, acc,
                                                   false, false);
    }
    const int n = nt * 16 + (lane & 15);
    const float bia = bias[n];
    const int g = lane >> 4;
#pragma unroll
    for (int r = 0; r < 8; ++r) {
      int row = mbase + r + 8 * g;
      float x = acc[r] + bia;
      if (mode == MODE_GELU) {
        float y = 0.5f * x * (1.0f + erff(x * 0.70710678118654752f));
        Ch[(size_t)row * N + n] = (_Float16)y;
      } else {  // MODE_RES
        Cf[(size_t)row * N + n] = x + residual[(size_t)row * N + n];
      }
    }
  }
}

// ---------------------------------------------------------------------------
// LayerNorm over rows of 256 (one block per row)
// ---------------------------------------------------------------------------
__global__ __launch_bounds__(256) void layernorm_k(
    const float* __restrict__ X, const float* __restrict__ g, const float* __restrict__ b,
    float* __restrict__ Yf, _Float16* __restrict__ Yh) {
  __shared__ float red[256];
  const int row = blockIdx.x, t = threadIdx.x;
  float x = X[(size_t)row * 256 + t];
  red[t] = x; __syncthreads();
  for (int s = 128; s > 0; s >>= 1) { if (t < s) red[t] += red[t + s]; __syncthreads(); }
  float mu = red[0] * (1.f / 256.f); __syncthreads();
  float d = x - mu;
  red[t] = d * d; __syncthreads();
  for (int s = 128; s > 0; s >>= 1) { if (t < s) red[t] += red[t + s]; __syncthreads(); }
  float var = red[0] * (1.f / 256.f);
  float y = d * rsqrtf(var + 1e-5f) * g[t] + b[t];
  if (Yf) Yf[(size_t)row * 256 + t] = y;
  if (Yh) Yh[(size_t)row * 256 + t] = (_Float16)y;
}

// ---------------------------------------------------------------------------
// host
// ---------------------------------------------------------------------------
extern "C" void kernel_launch(void* const* d_in, const int* in_sizes, int n_in,
                              void* d_out, int out_size, void* d_ws, size_t ws_size,
                              hipStream_t stream) {
  const float* h_n   = (const float*)d_in[0];
  const float* h_e   = (const float*)d_in[1];
  const int*   eidx  = (const int*)d_in[2];
  const float* Wq_w  = (const float*)d_in[3];
  const float* Wq_b  = (const float*)d_in[4];
  const float* Wkv_w = (const float*)d_in[5];
  const float* Wkv_b = (const float*)d_in[6];
  const float* Wo_w  = (const float*)d_in[7];
  const float* Wo_b  = (const float*)d_in[8];
  const float* ln1_g = (const float*)d_in[9];
  const float* ln1_b = (const float*)d_in[10];
  const float* f1_w  = (const float*)d_in[11];
  const float* f1_b  = (const float*)d_in[12];
  const float* f2_w  = (const float*)d_in[13];
  const float* f2_b  = (const float*)d_in[14];
  const float* ln2_g = (const float*)d_in[15];
  const float* ln2_b = (const float*)d_in[16];
  const int* srcIdx = eidx;
  const int* dstIdx = eidx + N_EDGES;
  float* out = (float*)d_out;

  char* w = (char*)d_ws;
  auto alloc = [&](size_t bytes) -> void* {
    void* p = (void*)w;
    w += (bytes + 255) & ~(size_t)255;
    return p;
  };
  _Float16* hn16   = (_Float16*)alloc((size_t)N_NODES * 256 * 2);
  _Float16* he16   = (_Float16*)alloc((size_t)N_EDGES * 256 * 2);
  float*    Wcat   = (float*)alloc((size_t)512 * 768 * 4);
  float*    b768   = (float*)alloc(768 * 4);
  _Float16* QKVp   = (_Float16*)alloc((size_t)512 * 768 * 2);
  _Float16* Wop    = (_Float16*)alloc((size_t)256 * 256 * 2);
  _Float16* F1p    = (_Float16*)alloc((size_t)256 * 1024 * 2);
  _Float16* F2p    = (_Float16*)alloc((size_t)1024 * 256 * 2);
  _Float16* Q16    = (_Float16*)alloc((size_t)N_EDGES * 256 * 2);
  _Float16* K16    = (_Float16*)alloc((size_t)N_EDGES * 256 * 2);
  _Float16* V16    = (_Float16*)alloc((size_t)N_EDGES * 256 * 2);
  float*    scores = (float*)alloc((size_t)N_EDGES * NHEAD * 4);
  float*    exB    = (float*)alloc((size_t)N_EDGES * NHEAD * 4);
  unsigned* mkey   = (unsigned*)alloc((size_t)N_NODES * NHEAD * 4);
  float*    denom  = (float*)alloc((size_t)N_NODES * NHEAD * 4);
  float*    agg    = (float*)alloc((size_t)N_NODES * 256 * 4);
  _Float16* agg16  = (_Float16*)alloc((size_t)N_NODES * 256 * 2);
  float*    y1     = (float*)alloc((size_t)N_NODES * 256 * 4);
  float*    h1     = (float*)alloc((size_t)N_NODES * 256 * 4);
  _Float16* h1_16  = (_Float16*)alloc((size_t)N_NODES * 256 * 2);
  _Float16* ff16   = (_Float16*)alloc((size_t)N_NODES * 1024 * 2);
  float*    y2     = (float*)alloc((size_t)N_NODES * 256 * 4);
  (void)ws_size; (void)n_in; (void)in_sizes; (void)out_size;

  auto blocks = [](long long n) { return (unsigned)((n + 255) / 256); };

  // casts + weight packing
  cast_f32_f16_k<<<blocks((long long)N_NODES * 256), 256, 0, stream>>>(h_n, hn16, (long long)N_NODES * 256);
  cast_f32_f16_k<<<blocks((long long)N_EDGES * 256), 256, 0, stream>>>(h_e, he16, (long long)N_EDGES * 256);
  concat_qkv_w_k<<<blocks(512 * 768), 256, 0, stream>>>(Wq_w, Wkv_w, Wcat);
  concat_bias_k<<<blocks(768), 256, 0, stream>>>(Wq_b, Wkv_b, b768);
  pack_b_k<<<blocks(512 * 768), 256, 0, stream>>>(Wcat, QKVp, 512, 768);
  pack_b_k<<<blocks(256 * 256), 256, 0, stream>>>(Wo_w, Wop, 256, 256);
  pack_b_k<<<blocks(256 * 1024), 256, 0, stream>>>(f1_w, F1p, 256, 1024);
  pack_b_k<<<blocks(1024 * 256), 256, 0, stream>>>(f2_w, F2p, 1024, 256);

  // fused per-edge QKV GEMM (WMMA f16, async-LDS staged A)
  edge_qkv_gemm_k<<<N_EDGES / 16, 256, 0, stream>>>(hn16, he16, srcIdx, dstIdx, QKVp, b768,
                                                    Q16, K16, V16);

  // segment softmax + aggregation
  hipMemsetAsync(mkey, 0, (size_t)N_NODES * NHEAD * 4, stream);   // key(0) <= every key
  hipMemsetAsync(denom, 0, (size_t)N_NODES * NHEAD * 4, stream);
  hipMemsetAsync(agg, 0, (size_t)N_NODES * 256 * 4, stream);
  scores_max_k<<<blocks((long long)N_EDGES * NHEAD), 256, 0, stream>>>(Q16, K16, srcIdx, scores, mkey);
  expsum_k<<<blocks((long long)N_EDGES * NHEAD), 256, 0, stream>>>(scores, mkey, srcIdx, exB, denom);
  agg_k<<<blocks((long long)N_EDGES * 256), 256, 0, stream>>>(exB, denom, srcIdx, V16, agg);

  // output projection + residual, LN1
  cast_f32_f16_k<<<blocks((long long)N_NODES * 256), 256, 0, stream>>>(agg, agg16, (long long)N_NODES * 256);
  node_gemm_k<<<N_NODES / 16, 256, 0, stream>>>(agg16, Wop, Wo_b, h_n, y1, nullptr,
                                                256, 256, MODE_RES);
  layernorm_k<<<N_NODES, 256, 0, stream>>>(y1, ln1_g, ln1_b, h1, h1_16);

  // FFN (GELU exact) + residual, LN2 -> output
  node_gemm_k<<<N_NODES / 16, 256, 0, stream>>>(h1_16, F1p, f1_b, nullptr, nullptr, ff16,
                                                1024, 256, MODE_GELU);
  node_gemm_k<<<N_NODES / 16, 256, 0, stream>>>(ff16, F2p, f2_b, h1, y2, nullptr,
                                                256, 1024, MODE_RES);
  layernorm_k<<<N_NODES, 256, 0, stream>>>(y2, ln2_g, ln2_b, out, nullptr);
}
